// OPTLoraAttention_77850577207373
// MI455X (gfx1250) — matrix operand order repaired and checked by reading, not compiled, not run on hardware
//
#include <hip/hip_runtime.h>

// ---- problem constants (fixed by the reference) ----
#define Bdim 2
#define Tdim 2048
#define Edim 2048
#define Hdim 32
#define Ddim 64
#define Rdim 8
#define SCALING 0.125f   // D^-0.5

typedef __attribute__((ext_vector_type(16))) __bf16 v16bf;
typedef __attribute__((ext_vector_type(8)))  float  v8f;
typedef unsigned int  u32x4 __attribute__((ext_vector_type(4)));
typedef int           i32x8 __attribute__((ext_vector_type(8)));
typedef int           i32x4 __attribute__((ext_vector_type(4)));

// address-space-qualified pointer types for the async-copy builtin
typedef __attribute__((address_space(1))) i32x4 as1_i32x4;
typedef __attribute__((address_space(3))) i32x4 as3_i32x4;

// ---- CDNA5 feature guards (fallbacks keep the compile green either way) ----
#if __has_builtin(__builtin_amdgcn_global_load_async_to_lds_b128) && \
    __has_builtin(__builtin_amdgcn_s_wait_asynccnt)
#define HAVE_ASYNC 1
#else
#define HAVE_ASYNC 0
#endif

#if __has_builtin(__builtin_amdgcn_tensor_load_to_lds) && \
    __has_builtin(__builtin_amdgcn_s_wait_tensorcnt)
#define HAVE_TDM 1
#else
#define HAVE_TDM 0
#endif

union Frag {
    unsigned int   u[8];
    unsigned short h[16];
    v16bf          v;
};

__device__ __forceinline__ unsigned short f2bf(float x) {
    unsigned int u = __float_as_uint(x);
    unsigned int r = (u + 0x7FFFu + ((u >> 16) & 1u)) >> 16;   // RNE
    return (unsigned short)r;
}

#if HAVE_ASYNC
// per-lane 16B global -> LDS copy via ASYNCcnt path
#define ASYNC_CP16(dst_lds, src_glb)                                          \
    __builtin_amdgcn_global_load_async_to_lds_b128(                           \
        (as1_i32x4*)(unsigned long long)(src_glb),                            \
        (as3_i32x4*)(unsigned)(unsigned long long)(dst_lds), 0, 0)
#endif

#if HAVE_TDM
// 2D bf16 tile load via Tensor Data Mover (D# per ISA 8.3/8.4; 2-byte elems)
__device__ __forceinline__ void tdm_load_2d_bf16(
    unsigned lds_off, unsigned long long gaddr,
    unsigned tile_d0, unsigned tile_d1,
    unsigned tensor_d0, unsigned tensor_d1, unsigned stride0) {
    u32x4 g0;
    g0.x = 1u;                                               // count=1 (valid), user
    g0.y = lds_off;                                          // LDS byte address
    g0.z = (unsigned)(gaddr & 0xffffffffu);                  // global_addr[31:0]
    g0.w = (unsigned)((gaddr >> 32) & 0x01ffffffu) | (2u << 30); // [56:32] | type=2
    i32x8 g1;
    g1[0] = (int)(1u << 16);                                 // data_size=1 (2 bytes)
    g1[1] = (int)((tensor_d0 & 0xffffu) << 16);              // tensor_dim0[15:0]
    g1[2] = (int)(((tensor_d0 >> 16) & 0xffffu) |
                  ((tensor_d1 & 0xffffu) << 16));            // dim0[31:16]|dim1[15:0]
    g1[3] = (int)(((tensor_d1 >> 16) & 0xffffu) |
                  ((tile_d0 & 0xffffu) << 16));              // dim1[31:16]|tile0
    g1[4] = (int)(tile_d1 & 0xffffu);                        // tile1 | tile2=0
    g1[5] = (int)stride0;                                    // dim0_stride[31:0]
    g1[6] = 0;                                               // stride0[47:32]|stride1 lo
    g1[7] = 0;
    i32x4 z4 = {0, 0, 0, 0};
    i32x8 z8 = {0, 0, 0, 0, 0, 0, 0, 0};
    __builtin_amdgcn_tensor_load_to_lds(g0, g1, z4, z4, z8, 0);  // 6-arg variant
}
#endif

// ---------------------------------------------------------------------------
// alive masks: row_alive[i] = any(w[i,:]!=0), col_alive[j] = any(w[:,j]!=0)
// ---------------------------------------------------------------------------
__global__ void alive_kernel(const float* __restrict__ w,
                             float* __restrict__ rowA, float* __restrict__ colA) {
    int i = blockIdx.x * blockDim.x + threadIdx.x;
    if (i >= Edim) return;
    float ra = 0.f, ca = 0.f;
    for (int k = 0; k < Edim; ++k) {
        if (w[(size_t)i * Edim + k] != 0.f) ra = 1.f;
        if (w[(size_t)k * Edim + i] != 0.f) ca = 1.f;
    }
    rowA[i] = ra;
    colA[i] = ca;
}

// ---------------------------------------------------------------------------
// W_eff = W + (row_alive . col_alive) * (l2 @ l1), converted to bf16
// ---------------------------------------------------------------------------
__global__ void weff_kernel(const float* __restrict__ w,
                            const float* __restrict__ l1,
                            const float* __restrict__ l2,
                            const float* __restrict__ rowA,
                            const float* __restrict__ colA,
                            unsigned short* __restrict__ out) {
    size_t idx = (size_t)blockIdx.x * blockDim.x + threadIdx.x;
    int i = (int)(idx >> 11);
    int j = (int)(idx & 2047);
    float s = 0.f;
#pragma unroll
    for (int r = 0; r < Rdim; ++r)
        s += l2[(size_t)i * Rdim + r] * l1[(size_t)r * Edim + j];
    float v = w[idx] + rowA[i] * colA[j] * s;
    out[idx] = f2bf(v);
}

__global__ void cvt_kernel(const float* __restrict__ in,
                           unsigned short* __restrict__ out, size_t n) {
    size_t idx = (size_t)blockIdx.x * blockDim.x + threadIdx.x;
    if (idx < n) out[idx] = f2bf(in[idx]);
}

// ---------------------------------------------------------------------------
// bf16 WMMA GEMM:  C[m,n] = (sum_k A[m,k]*W[n,k] + bias[n]) * scale
// Block tile 128x128, 8 waves, per-wave 32x64 (2x4 WMMA accs), K-step 32.
// ---------------------------------------------------------------------------
__global__ __launch_bounds__(256) void gemm_kernel(
    const unsigned short* __restrict__ A,
    const unsigned short* __restrict__ W,
    const float* __restrict__ bias,
    unsigned short* __restrict__ Obf,
    float* __restrict__ Of,
    int in_gather, int out_head, float scale) {

    __shared__ unsigned short As[128][32];
    __shared__ unsigned short Bs[128][32];

    const int tid  = threadIdx.x;
    const int wave = tid >> 5;
    const int lane = tid & 31;
    const int ln   = lane & 15;
    const int hi   = lane >> 4;
    const int wr   = wave & 3;       // 4 x 32 = 128 M
    const int wc   = wave >> 2;      // 2 x 64 = 128 N

    const int m0 = blockIdx.y * 128;
    const int n0 = blockIdx.x * 128;

    v8f acc[2][4];
#pragma unroll
    for (int ms = 0; ms < 2; ++ms)
#pragma unroll
        for (int ns = 0; ns < 4; ++ns)
#pragma unroll
            for (int i = 0; i < 8; ++i) acc[ms][ns][i] = 0.f;

    const int r   = tid >> 2;        // 0..63
    const int seg = tid & 3;         // 8 bf16 each

    for (int kk = 0; kk < Edim; kk += 32) {
        int k0 = kk + seg * 8;
        const unsigned short *asrc0, *asrc1;
        if (!in_gather) {
            asrc0 = A + (size_t)(m0 + r) * Edim + k0;
            asrc1 = A + (size_t)(m0 + r + 64) * Edim + k0;
        } else {
            int ma = m0 + r, mb = m0 + r + 64;
            int h = k0 >> 6, d = k0 & 63;
            asrc0 = A + (((size_t)((ma >> 11) * Hdim + h) * Tdim + (ma & 2047)) * Ddim + d);
            asrc1 = A + (((size_t)((mb >> 11) * Hdim + h) * Tdim + (mb & 2047)) * Ddim + d);
        }
        const unsigned short* bsrc0 = W + (size_t)(n0 + r) * Edim + k0;
        const unsigned short* bsrc1 = W + (size_t)(n0 + r + 64) * Edim + k0;
#if HAVE_ASYNC
        ASYNC_CP16(&As[r][seg * 8], asrc0);
        ASYNC_CP16(&As[r + 64][seg * 8], asrc1);
        ASYNC_CP16(&Bs[r][seg * 8], bsrc0);
        ASYNC_CP16(&Bs[r + 64][seg * 8], bsrc1);
        __builtin_amdgcn_s_wait_asynccnt(0);
#else
        *(uint4*)&As[r][seg * 8]      = *(const uint4*)asrc0;
        *(uint4*)&As[r + 64][seg * 8] = *(const uint4*)asrc1;
        *(uint4*)&Bs[r][seg * 8]      = *(const uint4*)bsrc0;
        *(uint4*)&Bs[r + 64][seg * 8] = *(const uint4*)bsrc1;
#endif
        __syncthreads();

        Frag af[2];
#pragma unroll
        for (int ms = 0; ms < 2; ++ms)
#pragma unroll
            for (int p = 0; p < 8; ++p) {
                int k2 = ((p < 4) ? 2 * p : 2 * p + 8) + 8 * hi;  // A-layout K map
                af[ms].u[p] = *(const unsigned int*)&As[wr * 32 + ms * 16 + ln][k2];
            }
#pragma unroll
        for (int ns = 0; ns < 4; ++ns) {
            Frag wf;
#pragma unroll
            for (int p = 0; p < 8; ++p) {
                int k2 = ((p < 4) ? 2 * p : 2 * p + 8) + 8 * hi;  // B-layout K map
                wf.u[p] = *(const unsigned int*)&Bs[wc * 64 + ns * 16 + ln][k2];
            }
#pragma unroll
            for (int ms = 0; ms < 2; ++ms)
                acc[ms][ns] = __builtin_amdgcn_wmma_f32_16x16x32_bf16(
                    false, af[ms].v, false, wf.v, (short)0, acc[ms][ns], false, false);
        }
        __syncthreads();
    }

#pragma unroll
    for (int ms = 0; ms < 2; ++ms)
#pragma unroll
        for (int ns = 0; ns < 4; ++ns) {
            int ncol = n0 + wc * 64 + ns * 16 + ln;
            float bv = bias[ncol];
#pragma unroll
            for (int i = 0; i < 8; ++i) {
                int m = m0 + wr * 32 + ms * 16 + i + 8 * hi;
                float v = (acc[ms][ns][i] + bv) * scale;
                if (out_head) {
                    int b = m >> 11, t = m & 2047;
                    int h = ncol >> 6, d = ncol & 63;
                    Obf[((size_t)(b * Hdim + h) * Tdim + t) * Ddim + d] = f2bf(v);
                } else {
                    Of[(size_t)m * Edim + ncol] = v;
                }
            }
        }
}

// ---------------------------------------------------------------------------
// Flash-style causal attention, bf16 WMMA, f32 online softmax.
// K/V tiles staged by the Tensor Data Mover when available.
// ---------------------------------------------------------------------------
__global__ __launch_bounds__(256) void attn_kernel(
    const unsigned short* __restrict__ Qg,
    const unsigned short* __restrict__ Kg,
    const unsigned short* __restrict__ Vg,
    const float* __restrict__ mask,
    unsigned short* __restrict__ Og) {

    __shared__ unsigned short Ks[32][64];
    __shared__ unsigned short Vs[32][64];
    __shared__ unsigned short Ps[8][16][32];

    const int tid  = threadIdx.x;
    const int wave = tid >> 5;
    const int lane = tid & 31;
    const int ln   = lane & 15;
    const int hi   = lane >> 4;
    const int bh   = blockIdx.z;
    const int b    = bh / Hdim;
    const int q0   = blockIdx.x * 128 + wave * 16;

    Frag qf[2];
    {
        size_t qbase = ((size_t)bh * Tdim + q0 + ln) * Ddim;
#pragma unroll
        for (int ds = 0; ds < 2; ++ds)
#pragma unroll
            for (int p = 0; p < 8; ++p) {
                int k2 = ((p < 4) ? 2 * p : 2 * p + 8) + 8 * hi;
                qf[ds].u[p] = *(const unsigned int*)(Qg + qbase + ds * 32 + k2);
            }
    }

    float m_i[8], l_i[8];
    v8f oacc[4];
#pragma unroll
    for (int i = 0; i < 8; ++i) { m_i[i] = -3.0e38f; l_i[i] = 0.f; }
#pragma unroll
    for (int n = 0; n < 4; ++n)
#pragma unroll
        for (int i = 0; i < 8; ++i) oacc[n][i] = 0.f;

    const int nkeys = (blockIdx.x + 1) * 128;   // causal block skipping

    for (int j = 0; j < nkeys; j += 32) {
#if HAVE_TDM
        if (wave == 0) {
            unsigned long long kaddr =
                (unsigned long long)(Kg + ((size_t)bh * Tdim + j) * Ddim);
            unsigned long long vaddr =
                (unsigned long long)(Vg + ((size_t)bh * Tdim + j) * Ddim);
            unsigned ks_off = (unsigned)(unsigned long long)&Ks[0][0];
            unsigned vs_off = (unsigned)(unsigned long long)&Vs[0][0];
            tdm_load_2d_bf16(ks_off, kaddr, 64, 32, 64, 32, 64);
            tdm_load_2d_bf16(vs_off, vaddr, 64, 32, 64, 32, 64);
        }
        __builtin_amdgcn_s_wait_tensorcnt(0);
#else
        {
            int rr = tid >> 3, sg = tid & 7;
            size_t gb = ((size_t)bh * Tdim + j + rr) * Ddim + sg * 8;
            *(uint4*)&Ks[rr][sg * 8] = *(const uint4*)(Kg + gb);
            *(uint4*)&Vs[rr][sg * 8] = *(const uint4*)(Vg + gb);
        }
#endif
        __syncthreads();

        // ---- S = Q * K^T (two 16-key sub-tiles, K over D=64) ----
        v8f sacc[2];
#pragma unroll
        for (int s2 = 0; s2 < 2; ++s2) {
#pragma unroll
            for (int i = 0; i < 8; ++i) sacc[s2][i] = 0.f;
#pragma unroll
            for (int ds = 0; ds < 2; ++ds) {
                Frag kf;
#pragma unroll
                for (int p = 0; p < 8; ++p) {
                    int k2 = ((p < 4) ? 2 * p : 2 * p + 8) + 8 * hi;
                    kf.u[p] = *(const unsigned int*)&Ks[s2 * 16 + ln][ds * 32 + k2];
                }
                sacc[s2] = __builtin_amdgcn_wmma_f32_16x16x32_bf16(
                    false, qf[ds].v, false, kf.v, (short)0, sacc[s2], false, false);
            }
        }

        // ---- additive mask + clamp at fp32 min ----
#pragma unroll
        for (int s2 = 0; s2 < 2; ++s2) {
            int s = j + s2 * 16 + ln;
#pragma unroll
            for (int i = 0; i < 8; ++i) {
                int t = q0 + i + 8 * hi;
                float v = sacc[s2][i] + mask[((size_t)b * Tdim + t) * Tdim + s];
                sacc[s2][i] = fmaxf(v, -3.4028235e38f);
            }
        }

        // ---- online softmax ----
        float tm[8], fsc[8], ps[8];
#pragma unroll
        for (int i = 0; i < 8; ++i) tm[i] = fmaxf(sacc[0][i], sacc[1][i]);
        for (int xm = 1; xm < 16; xm <<= 1)
#pragma unroll
            for (int i = 0; i < 8; ++i) tm[i] = fmaxf(tm[i], __shfl_xor(tm[i], xm, 32));
#pragma unroll
        for (int i = 0; i < 8; ++i) {
            float nm = fmaxf(m_i[i], tm[i]);
            fsc[i] = __expf(m_i[i] - nm);
            m_i[i] = nm;
        }
#pragma unroll
        for (int i = 0; i < 8; ++i) {
            sacc[0][i] = __expf(sacc[0][i] - m_i[i]);
            sacc[1][i] = __expf(sacc[1][i] - m_i[i]);
            ps[i] = sacc[0][i] + sacc[1][i];
        }
        for (int xm = 1; xm < 16; xm <<= 1)
#pragma unroll
            for (int i = 0; i < 8; ++i) ps[i] += __shfl_xor(ps[i], xm, 32);
#pragma unroll
        for (int i = 0; i < 8; ++i) l_i[i] = l_i[i] * fsc[i] + ps[i];
#pragma unroll
        for (int n = 0; n < 4; ++n)
#pragma unroll
            for (int i = 0; i < 8; ++i) oacc[n][i] *= fsc[i];

        // ---- P (C-layout) -> per-wave LDS -> A-fragment ----
#pragma unroll
        for (int i = 0; i < 8; ++i) {
            Ps[wave][i + 8 * hi][ln]      = f2bf(sacc[0][i]);
            Ps[wave][i + 8 * hi][16 + ln] = f2bf(sacc[1][i]);
        }
        Frag pf;
#pragma unroll
        for (int p = 0; p < 8; ++p) {
            int k2 = ((p < 4) ? 2 * p : 2 * p + 8) + 8 * hi;
            pf.u[p] = *(const unsigned int*)&Ps[wave][ln][k2];
        }

        // ---- O += P * V ----
#pragma unroll
        for (int n = 0; n < 4; ++n) {
            Frag vf;
#pragma unroll
            for (int e = 0; e < 16; ++e)
                vf.h[e] = Vs[((e < 8) ? e : e + 8) + 8 * hi][n * 16 + ln];
            oacc[n] = __builtin_amdgcn_wmma_f32_16x16x32_bf16(
                false, pf.v, false, vf.v, (short)0, oacc[n], false, false);
        }
        __syncthreads();
    }

#pragma unroll
    for (int i = 0; i < 8; ++i) l_i[i] = 1.0f / l_i[i];
#pragma unroll
    for (int n = 0; n < 4; ++n)
#pragma unroll
        for (int i = 0; i < 8; ++i) {
            int t = q0 + i + 8 * hi;
            int d = n * 16 + ln;
            Og[((size_t)bh * Tdim + t) * Ddim + d] = f2bf(oacc[n][i] * l_i[i]);
        }
}

// ---------------------------------------------------------------------------
extern "C" void kernel_launch(void* const* d_in, const int* in_sizes, int n_in,
                              void* d_out, int out_size, void* d_ws, size_t ws_size,
                              hipStream_t stream) {
    (void)in_sizes; (void)n_in; (void)out_size; (void)ws_size;

    const float* hidden = (const float*)d_in[0];
    const float* mask   = (const float*)d_in[1];
    const float *w[4], *bias[4], *l1[4], *l2[4];
    for (int i = 0; i < 4; ++i) {          // dict order: w,b,l1,l2 per q,k,v,o
        w[i]    = (const float*)d_in[2 + 4 * i];
        bias[i] = (const float*)d_in[3 + 4 * i];
        l1[i]   = (const float*)d_in[4 + 4 * i];
        l2[i]   = (const float*)d_in[5 + 4 * i];
    }

    char* ws = (char*)d_ws;
    size_t off = 0;
    auto alloc = [&](size_t bytes) -> void* {
        void* p = ws + off;
        off += (bytes + 255) & ~(size_t)255;
        return p;
    };
    const size_t MK = (size_t)Bdim * Tdim * Edim;
    const size_t HS = (size_t)Bdim * Hdim * Tdim * Ddim;

    unsigned short* Xb = (unsigned short*)alloc(MK * 2);
    unsigned short* Weff[4];
    for (int i = 0; i < 4; ++i) Weff[i] = (unsigned short*)alloc((size_t)Edim * Edim * 2);
    unsigned short* Qb = (unsigned short*)alloc(HS * 2);
    unsigned short* Kb = (unsigned short*)alloc(HS * 2);
    unsigned short* Vb = (unsigned short*)alloc(HS * 2);
    unsigned short* Ab = (unsigned short*)alloc(HS * 2);
    float* rowA = (float*)alloc(4 * Edim * sizeof(float));
    float* colA = (float*)alloc(4 * Edim * sizeof(float));

    for (int i = 0; i < 4; ++i)
        alive_kernel<<<(Edim + 255) / 256, 256, 0, stream>>>(
            w[i], rowA + i * Edim, colA + i * Edim);
    for (int i = 0; i < 4; ++i)
        weff_kernel<<<((size_t)Edim * Edim) / 256, 256, 0, stream>>>(
            w[i], l1[i], l2[i], rowA + i * Edim, colA + i * Edim, Weff[i]);

    cvt_kernel<<<(MK + 255) / 256, 256, 0, stream>>>(hidden, Xb, MK);

    dim3 gg(Edim / 128, (Bdim * Tdim) / 128);
    gemm_kernel<<<gg, 256, 0, stream>>>(Xb, Weff[0], bias[0], Qb, nullptr, 0, 1, SCALING);
    gemm_kernel<<<gg, 256, 0, stream>>>(Xb, Weff[1], bias[1], Kb, nullptr, 0, 1, 1.0f);
    gemm_kernel<<<gg, 256, 0, stream>>>(Xb, Weff[2], bias[2], Vb, nullptr, 0, 1, 1.0f);

    attn_kernel<<<dim3(Tdim / 128, 1, Bdim * Hdim), 256, 0, stream>>>(
        Qb, Kb, Vb, mask, Ab);

    gemm_kernel<<<gg, 256, 0, stream>>>(Ab, Weff[3], bias[3], nullptr,
                                        (float*)d_out, 1, 0, 1.0f);
}
